// Quanvolution__gen329_65481071408268
// MI455X (gfx1250) — matrix operand order repaired and checked by reading, not compiled, not run on hardware
//
#include <hip/hip_runtime.h>
#include <hip/hip_bf16.h>
#include <math.h>

typedef _Float16 h16;
typedef __attribute__((ext_vector_type(4)))  _Float16 v4h;
typedef __attribute__((ext_vector_type(8)))  _Float16 v8h;
typedef __attribute__((ext_vector_type(16))) _Float16 v16h;
typedef __attribute__((ext_vector_type(8)))  float    v8f;

#define B_IMG   1024
#define PATCHES 196            // 14*14
#define NCIRC   (B_IMG * PATCHES)
#define KDIM    784
#define KPAD    800            // 784 padded to multiple of 32
#define NCLS    10

// workspace layout (bytes)
#define QH_OFF   0                              // f16 [1024][800]  = 1,638,400 B
#define WHP_OFF  (B_IMG * KPAD * 2)             // f16 [16][800]    =    25,600 B
#define LOG_OFF  (WHP_OFF + 16 * KPAD * 2)      // f32 [1024][16]   =    65,536 B

// ---------------------------------------------------------------------------
// Pad / weight-conversion kernel: zero qh K-pad columns, build padded f16 W.
// whp[n][k] = (n<10 && k<784) ? W[n][k] : 0   (B-matrix source: B[k][n]=W[n][k])
// ---------------------------------------------------------------------------
__global__ void pad_init_kernel(const float* __restrict__ W,
                                h16* __restrict__ qh, h16* __restrict__ whp) {
    int t = blockIdx.x * blockDim.x + threadIdx.x;
    const int qpad = B_IMG * 16;
    if (t < qpad) {
        int row = t >> 4, c = t & 15;
        qh[row * KPAD + KDIM + c] = (h16)0.f;
    } else {
        t -= qpad;
        if (t < 16 * KPAD) {
            int n = t / KPAD, k = t - n * KPAD;
            float v = (n < NCLS && k < KDIM) ? W[n * KDIM + k] : 0.f;
            whp[t] = (h16)v;
        }
    }
}

// ---------------------------------------------------------------------------
// 4-qubit state-vector simulation, one thread per 2x2 patch.
// State index bit (3-w) corresponds to wire w (wire 0 = MSB).
// ---------------------------------------------------------------------------
__device__ __forceinline__ void apply_ry(float* sr, float* si, int w, float c, float s) {
    const int st = 1 << (3 - w);
    #pragma unroll
    for (int i = 0; i < 16; ++i) {
        if (i & st) continue;
        const int j = i | st;
        float a0r = sr[i], a0i = si[i], a1r = sr[j], a1i = si[j];
        sr[i] = c * a0r - s * a1r;  si[i] = c * a0i - s * a1i;
        sr[j] = s * a0r + c * a1r;  si[j] = s * a0i + c * a1i;
    }
}

__device__ __forceinline__ void apply1q(float* sr, float* si, int w,
                                        float m00r, float m00i, float m01r, float m01i,
                                        float m10r, float m10i, float m11r, float m11i) {
    const int st = 1 << (3 - w);
    #pragma unroll
    for (int i = 0; i < 16; ++i) {
        if (i & st) continue;
        const int j = i | st;
        float a0r = sr[i], a0i = si[i], a1r = sr[j], a1i = si[j];
        sr[i] = m00r * a0r - m00i * a0i + m01r * a1r - m01i * a1i;
        si[i] = m00r * a0i + m00i * a0r + m01r * a1i + m01i * a1r;
        sr[j] = m10r * a0r - m10i * a0i + m11r * a1r - m11i * a1i;
        si[j] = m10r * a0i + m10i * a0r + m11r * a1i + m11i * a1r;
    }
}

__device__ __forceinline__ void apply_cnot(float* sr, float* si, int c, int t) {
    const int cb = 1 << (3 - c), tb = 1 << (3 - t);
    #pragma unroll
    for (int i = 0; i < 16; ++i) {
        if ((i & cb) && !(i & tb)) {
            const int j = i | tb;
            float tr = sr[i]; sr[i] = sr[j]; sr[j] = tr;
            float ti = si[i]; si[i] = si[j]; si[j] = ti;
        }
    }
}

__global__ void quanv_sim_kernel(const float* __restrict__ x,
                                 const float* __restrict__ params,
                                 h16* __restrict__ qh) {
    int t = blockIdx.x * blockDim.x + threadIdx.x;
    if (t >= NCIRC) return;
    int b  = t / PATCHES;
    int pi = t - b * PATCHES;
    int i  = pi / 14;
    int j  = pi - i * 14;

    const float* xb = x + b * 784 + (2 * i) * 28 + 2 * j;
    float ang[4] = { xb[0], xb[1], xb[28], xb[29] };   // wires 0..3

    float sr[16], si[16];
    #pragma unroll
    for (int k = 0; k < 16; ++k) { sr[k] = 0.f; si[k] = 0.f; }
    sr[0] = 1.f;

    // data encoding: RY(angle_w) on wire w
    #pragma unroll
    for (int w = 0; w < 4; ++w) {
        float s, c;
        sincosf(ang[w] * 0.5f, &s, &c);
        apply_ry(sr, si, w, c, s);
    }

    // StronglyEntanglingLayers: Rot(phi,theta,omega) then ring CNOTs with range l+1
    #pragma unroll
    for (int l = 0; l < 3; ++l) {
        #pragma unroll
        for (int w = 0; w < 4; ++w) {
            float phi = params[(l * 4 + w) * 3 + 0];
            float th  = params[(l * 4 + w) * 3 + 1];
            float om  = params[(l * 4 + w) * 3 + 2];
            float stt, ct, sp, cp, sm, cm;
            sincosf(th * 0.5f, &stt, &ct);
            sincosf((phi + om) * 0.5f, &sp, &cp);
            sincosf((phi - om) * 0.5f, &sm, &cm);
            // m00 = e^{-i(phi+om)/2} c ; m01 = -e^{+i(phi-om)/2} s
            // m10 = e^{-i(phi-om)/2} s ; m11 = e^{+i(phi+om)/2} c
            apply1q(sr, si, w,
                    cp * ct, -sp * ct,
                   -cm * stt, -sm * stt,
                    cm * stt, -sm * stt,
                    cp * ct,  sp * ct);
        }
        const int r = l + 1;   // l % (N_WIRES-1) + 1
        #pragma unroll
        for (int w = 0; w < 4; ++w)
            apply_cnot(sr, si, w, (w + r) & 3);
    }

    // <Z_w> = sum_idx prob[idx] * (bit(3-w)==0 ? +1 : -1)
    float z0 = 0.f, z1 = 0.f, z2 = 0.f, z3 = 0.f;
    #pragma unroll
    for (int k = 0; k < 16; ++k) {
        float p = sr[k] * sr[k] + si[k] * si[k];
        z0 += (k & 8) ? -p : p;
        z1 += (k & 4) ? -p : p;
        z2 += (k & 2) ? -p : p;
        z3 += (k & 1) ? -p : p;
    }

    v4h q;
    q[0] = (h16)z0; q[1] = (h16)z1; q[2] = (h16)z2; q[3] = (h16)z3;
    *(v4h*)(qh + b * KPAD + pi * 4) = q;   // 8-byte aligned store
}

// ---------------------------------------------------------------------------
// logits[1024][16] = qh[1024][800] @ whp[16][800]^T via v_wmma_f32_16x16x32_f16
// One wave (32 threads) per 16-row tile; 25 k-steps of 32.
// Fragment layouts as loaded here (wave32; contiguous k, not the ISA table order):
//   A 16x32 f16 : lane L holds row (L&15); halves 0..15 = K (L>>4)*16 .. +15
//   B 32x16 f16 : lane L holds col (L&15); halves 0..15 = K (L>>4)*16 .. +15
//   C/D 16x16 f32: lane L holds col (L&15); vgpr v = row (L>>4)*8 + v
// ---------------------------------------------------------------------------
__global__ void wmma_gemm_kernel(const h16* __restrict__ qh,
                                 const h16* __restrict__ whp,
                                 float* __restrict__ logits) {
    const int lane = threadIdx.x;        // 0..31, full wave, EXEC all-1s
    const int row0 = blockIdx.x * 16;
    const int r    = lane & 15;
    const int hi   = lane >> 4;

    const h16* arow = qh  + (row0 + r) * KPAD;  // A row for this lane
    const h16* bcol = whp + r * KPAD;           // B column for this lane

    v8f acc = {};
    for (int k0 = 0; k0 < KPAD; k0 += 32) {
        v8h alo = *(const v8h*)(arow + k0 + hi * 16);       // K hi*16..hi*16+7
        v8h ahi = *(const v8h*)(arow + k0 + hi * 16 + 8);   // K hi*16+8..hi*16+15
        v16h a;
        #pragma unroll
        for (int u = 0; u < 8; ++u) { a[u] = alo[u]; a[8 + u] = ahi[u]; }

        v16h bf = *(const v16h*)(bcol + k0 + hi * 16);      // 32B aligned

        acc = __builtin_amdgcn_wmma_f32_16x16x32_f16(
                  false, a, false, bf, (short)0, acc, false, false);
    }

    #pragma unroll
    for (int v = 0; v < 8; ++v) {
        int row = row0 + hi * 8 + v;
        logits[row * 16 + r] = acc[v];
    }
}

// ---------------------------------------------------------------------------
// log_softmax over the 10 real classes, one thread per image row.
// ---------------------------------------------------------------------------
__global__ void logsoftmax_kernel(const float* __restrict__ logits,
                                  const float* __restrict__ bias,
                                  float* __restrict__ out) {
    int row = blockIdx.x * blockDim.x + threadIdx.x;
    if (row >= B_IMG) return;
    float v[NCLS];
    float m = -3.0e38f;
    #pragma unroll
    for (int n = 0; n < NCLS; ++n) {
        v[n] = logits[row * 16 + n] + bias[n];
        m = fmaxf(m, v[n]);
    }
    float s = 0.f;
    #pragma unroll
    for (int n = 0; n < NCLS; ++n) s += expf(v[n] - m);
    float ls = logf(s);
    #pragma unroll
    for (int n = 0; n < NCLS; ++n) out[row * NCLS + n] = v[n] - m - ls;
}

// ---------------------------------------------------------------------------
extern "C" void kernel_launch(void* const* d_in, const int* in_sizes, int n_in,
                              void* d_out, int out_size, void* d_ws, size_t ws_size,
                              hipStream_t stream) {
    const float* x      = (const float*)d_in[0];   // [1024,1,28,28]
    const float* params = (const float*)d_in[1];   // [3,4,3]
    const float* W      = (const float*)d_in[2];   // [10,784]
    const float* bias   = (const float*)d_in[3];   // [10]
    float* out          = (float*)d_out;           // [1024,10]

    char*  ws     = (char*)d_ws;
    h16*   qh     = (h16*)(ws + QH_OFF);
    h16*   whp    = (h16*)(ws + WHP_OFF);
    float* logits = (float*)(ws + LOG_OFF);

    {
        const int total = B_IMG * 16 + 16 * KPAD;
        pad_init_kernel<<<(total + 255) / 256, 256, 0, stream>>>(W, qh, whp);
    }
    quanv_sim_kernel<<<(NCIRC + 255) / 256, 256, 0, stream>>>(x, params, qh);
    wmma_gemm_kernel<<<B_IMG / 16, 32, 0, stream>>>(qh, whp, logits);
    logsoftmax_kernel<<<(B_IMG + 255) / 256, 256, 0, stream>>>(logits, bias, out);
}
